// GQMultiHeadAttention_81501299409087
// MI455X (gfx1250) — matrix operand order repaired and checked by reading, not compiled, not run on hardware
//
#include <hip/hip_runtime.h>
#include <cstdint>
#include <cstddef>

// ---------------------------------------------------------------------------
// GQA attention forward for MI455X (gfx1250), bf16 WMMA datapath.
//   b=2, s=2048, d=2048, 32 q-heads / 8 kv-heads, head_dim=64, causal+RoPE.
// Pipeline: cvt/transposes -> QKV GEMMs (WMMA bf16, async-LDS double buffer,
//           K-step 64, bank-conflict-free padded LDS) -> RoPE (q pre-scaled)
//           -> flash attention (WMMA bf16, online softmax w/ DPP8+permlane16)
//           -> out GEMM (WMMA bf16, f32 out).
// ---------------------------------------------------------------------------

using bf16 = __bf16;
typedef bf16  bf16x16 __attribute__((ext_vector_type(16)));
typedef float f32x8   __attribute__((ext_vector_type(8)));

constexpr int Bc  = 2;
constexpr int Sc  = 2048;
constexpr int Dc  = 2048;
constexpr int Hc  = 32;
constexpr int HKV = 8;
constexpr int HD  = 64;
constexpr int DKV = HKV * HD;   // 512

constexpr int LDSP = 72;        // padded LDS row (64 data + 8 pad bf16)
                                // row stride 144B = 36 banks -> 16 consecutive
                                // rows map to distinct bank groups (36k mod 64)

union AB {
  bf16x16 v;
  uint4   u4[2];
};

__device__ inline uint4 ld4g(const bf16* p) {
  return *reinterpret_cast<const uint4*>(p);
}

__device__ inline f32x8 zero8() {
  f32x8 z;
#pragma unroll
  for (int e = 0; e < 8; ++e) z[e] = 0.0f;
  return z;
}

__device__ inline f32x8 wmma_bf16(bf16x16 a, bf16x16 b, f32x8 c) {
#if defined(__HIP_DEVICE_COMPILE__) && defined(__gfx1250__)
  // (neg_a, A, neg_b, B, c_mod, C, reuse_a, reuse_b)
  return __builtin_amdgcn_wmma_f32_16x16x32_bf16(false, a, false, b,
                                                 (short)0, c, false, false);
#else
  (void)a; (void)b;
  return c;  // host-pass stub, never executed
#endif
}

// ---------------------------------------------------------------------------
// Async global->LDS copy (ASYNCcnt-tracked, CDNA5 §4 async memory ops).
// ---------------------------------------------------------------------------
__device__ inline uint32_t lds_off(const void* p) {
#if defined(__HIP_DEVICE_COMPILE__)
  return (uint32_t)(uintptr_t)(__attribute__((address_space(3))) const void*)p;
#else
  (void)p; return 0;
#endif
}

__device__ inline void async_ld_b128(uint32_t ldsOff, const void* gp) {
#if defined(__HIP_DEVICE_COMPILE__) && defined(__gfx1250__)
  asm volatile("global_load_async_to_lds_b128 %0, %1, off"
               :: "v"(ldsOff), "v"(gp) : "memory");
#else
  (void)ldsOff; (void)gp;
#endif
}

__device__ inline void wait_async0() {
#if defined(__HIP_DEVICE_COMPILE__) && defined(__gfx1250__)
  asm volatile("s_wait_asynccnt 0x0" ::: "memory");
#endif
}

// ---------------------------------------------------------------------------
// 16-lane row reductions on the WMMA C-layout (rows live across 16 lanes).
// xor-1/2/4 via v_mov_b32 dpp8, xor-8 via v_permlane16_b32 (pure VALU; the
// __shfl_xor fallback lowers to ds_bpermute which burns LDS bandwidth).
// ---------------------------------------------------------------------------
#if defined(__HIP_DEVICE_COMPILE__) && defined(__gfx1250__) &&                 \
    __has_builtin(__builtin_amdgcn_mov_dpp8) &&                                \
    __has_builtin(__builtin_amdgcn_permlane16)
#define USE_DPP_RED 1
#endif

__device__ inline float red_max16(float x) {
#ifdef USE_DPP_RED
  // dpp8 selectors: lane i <- lane (i^k) within groups of 8
  x = fmaxf(x, __uint_as_float(__builtin_amdgcn_mov_dpp8(__float_as_uint(x), 14570689)));  // ^1
  x = fmaxf(x, __uint_as_float(__builtin_amdgcn_mov_dpp8(__float_as_uint(x), 11788826)));  // ^2
  x = fmaxf(x, __uint_as_float(__builtin_amdgcn_mov_dpp8(__float_as_uint(x),  6852524)));  // ^4
  {
    uint32_t xi = __float_as_uint(x);  // lane i <- lane (i^8) within rows of 16
    x = fmaxf(x, __uint_as_float(__builtin_amdgcn_permlane16(
                     xi, xi, 0xFEDCBA98u, 0x76543210u, false, false)));
  }
  return x;
#else
  for (int off = 1; off < 16; off <<= 1) x = fmaxf(x, __shfl_xor(x, off, 32));
  return x;
#endif
}

__device__ inline float red_sum16(float x) {
#ifdef USE_DPP_RED
  x += __uint_as_float(__builtin_amdgcn_mov_dpp8(__float_as_uint(x), 14570689));
  x += __uint_as_float(__builtin_amdgcn_mov_dpp8(__float_as_uint(x), 11788826));
  x += __uint_as_float(__builtin_amdgcn_mov_dpp8(__float_as_uint(x),  6852524));
  {
    uint32_t xi = __float_as_uint(x);
    x += __uint_as_float(__builtin_amdgcn_permlane16(
             xi, xi, 0xFEDCBA98u, 0x76543210u, false, false));
  }
  return x;
#else
  for (int off = 1; off < 16; off <<= 1) x += __shfl_xor(x, off, 32);
  return x;
#endif
}

// ---------------------------------------------------------------------------
// Elementwise f32 -> bf16 conversion (x activations).
// ---------------------------------------------------------------------------
__global__ void cvt_kernel(const float* __restrict__ src, bf16* __restrict__ dst,
                           size_t n) {
  size_t i = (size_t)blockIdx.x * blockDim.x + threadIdx.x;
  const size_t stride = (size_t)gridDim.x * blockDim.x;
  for (; i < n; i += stride) dst[i] = (bf16)src[i];
}

// W (K x N) f32 row-major -> Wt (N x K) bf16 row-major (one-time weight prep,
// makes WMMA B-fragments K-contiguous -> two b128 loads per lane).
__global__ void transpose_cvt_kernel(const float* __restrict__ W,
                                     bf16* __restrict__ Wt, int K, int N) {
  size_t idx = (size_t)blockIdx.x * blockDim.x + threadIdx.x;
  if (idx >= (size_t)K * N) return;
  const int k = (int)(idx / N);
  const int n = (int)(idx % N);
  Wt[(size_t)n * K + k] = (bf16)W[idx];
}

// ---------------------------------------------------------------------------
// WMMA bf16 GEMM with async-LDS double buffering, K-step 64 per buffer.
// C[M,N] = A[M,K] @ W[K,N], W given transposed (N x K).
// Block = 256 threads = 8 waves (4 M x 2 N), block tile 128x64.
// Per pipeline stage: 16 KB A-tile + 8 KB B-tile stream into LDS via
// global_load_async_to_lds_b128 (6 per thread) while each wave runs 8 WMMAs
// on the other buffer -> one barrier + one s_wait_asynccnt per 64 K-depth.
// LDS rows padded to 144B so ds_load_b128 fragment reads are conflict-free.
// MODE 0: bf16 row-major store. MODE 1: f32 row-major store.
// MODE 2: bf16 store transposed per kv-head: vT[(b*HKV+hk)*64+dd][s].
// ---------------------------------------------------------------------------
template <int MODE>
__global__ __launch_bounds__(256, 2) void gemm_kernel(
    const bf16* __restrict__ A, const bf16* __restrict__ Wt,
    void* __restrict__ Cout, int M, int N, int K) {
  __shared__ __align__(16) bf16 As[2][128][LDSP];  // 36 KB
  __shared__ __align__(16) bf16 Bs[2][64][LDSP];   // 18 KB

  const int t    = threadIdx.x;
  const int lane = t & 31;
  const int w    = t >> 5;
  const int ln   = lane & 15;   // row/col within a 16-wide fragment
  const int g    = lane >> 4;   // half-wave group (K-slice selector)
  const int wm   = w & 3, wn = w >> 2;
  const int tileMblk = blockIdx.x * 128;
  const int tileNblk = blockIdx.y * 64;

  auto issue_copy = [&](int buf, int kk) {
    // A tile: 128 rows x 64 cols bf16 = 1024 x 16B chunks, 4 per thread.
#pragma unroll
    for (int i = 0; i < 4; ++i) {
      const int c    = t + i * 256;
      const int row  = c >> 3;
      const int coff = (c & 7) * 8;
      async_ld_b128(lds_off(&As[buf][row][coff]),
                    A + (size_t)(tileMblk + row) * K + kk + coff);
    }
    // B tile: 64 rows x 64 cols = 512 chunks, 2 per thread.
#pragma unroll
    for (int i = 0; i < 2; ++i) {
      const int c    = t + i * 256;
      const int row  = c >> 3;
      const int coff = (c & 7) * 8;
      async_ld_b128(lds_off(&Bs[buf][row][coff]),
                    Wt + (size_t)(tileNblk + row) * K + kk + coff);
    }
  };

  f32x8 acc[2][2];
#pragma unroll
  for (int i = 0; i < 2; ++i)
#pragma unroll
    for (int j = 0; j < 2; ++j) acc[i][j] = zero8();

  issue_copy(0, 0);

  for (int kk = 0; kk < K; kk += 64) {
    const int buf = (kk >> 6) & 1;
    wait_async0();      // this wave's copies into `buf` have landed
    __syncthreads();    // everyone's copies landed; prior reads of buf done
    if (kk + 64 < K) issue_copy(buf ^ 1, kk + 64);

#pragma unroll
    for (int ks = 0; ks < 64; ks += 32) {
      AB a[2], b[2];
#pragma unroll
      for (int i = 0; i < 2; ++i) {
        const bf16* p = &As[buf][wm * 32 + i * 16 + ln][ks];
        a[i].u4[0] = *reinterpret_cast<const uint4*>(p + g * 16);
        a[i].u4[1] = *reinterpret_cast<const uint4*>(p + g * 16 + 8);
      }
#pragma unroll
      for (int j = 0; j < 2; ++j) {
        const bf16* p = &Bs[buf][wn * 32 + j * 16 + ln][ks];
        b[j].u4[0] = *reinterpret_cast<const uint4*>(p + g * 16);
        b[j].u4[1] = *reinterpret_cast<const uint4*>(p + g * 16 + 8);
      }
#pragma unroll
      for (int i = 0; i < 2; ++i)
#pragma unroll
        for (int j = 0; j < 2; ++j)
          acc[i][j] = wmma_bf16(a[i].v, b[j].v, acc[i][j]);
    }
  }

  // C/D layout: element r -> row = r + 8*g, col = ln
#pragma unroll
  for (int i = 0; i < 2; ++i)
#pragma unroll
    for (int j = 0; j < 2; ++j)
#pragma unroll
      for (int r = 0; r < 8; ++r) {
        const int m = tileMblk + wm * 32 + i * 16 + r + 8 * g;
        const int n = tileNblk + wn * 32 + j * 16 + ln;
        const float val = acc[i][j][r];
        if (MODE == 0) {
          ((bf16*)Cout)[(size_t)m * N + n] = (bf16)val;
        } else if (MODE == 1) {
          ((float*)Cout)[(size_t)m * N + n] = val;
        } else {  // MODE 2: V transposed per kv-head (N == DKV)
          const int bI = m >> 11;         // / Sc
          const int s  = m & (Sc - 1);
          const int hk = n >> 6;
          const int dd = n & 63;
          ((bf16*)Cout)[(((size_t)bI * HKV + hk) * HD + dd) * Sc + s] =
              (bf16)val;
        }
      }
}

// ---------------------------------------------------------------------------
// In-place RoPE on (B, S, nheads, 64) bf16, with output scale (the attention
// 1/sqrt(hd) is folded into q here). One thread per (b,s,h,i), i<32.
// ---------------------------------------------------------------------------
__global__ void rope_kernel(bf16* __restrict__ buf, int nheads, float scale) {
  size_t idx = (size_t)blockIdx.x * blockDim.x + threadIdx.x;
  const size_t total = (size_t)Bc * Sc * nheads * 32;
  if (idx >= total) return;
  const int i = (int)(idx & 31);
  size_t t = idx >> 5;
  const int h = (int)(t % nheads); t /= nheads;
  const int s = (int)(t % Sc);
  const int b = (int)(t / Sc);
  // inv_freq = 10000^(-i/32)  ->  exp(-i * ln(10000)/32)
  const float inv = __expf(-(float)i * 0.28782313662425572f);
  const float fr  = (float)s * inv;
  const float c = __cosf(fr), sn = __sinf(fr);
  bf16* p = buf + (((size_t)b * Sc + s) * nheads + h) * HD;
  const float t1 = (float)p[i];
  const float t2 = (float)p[i + 32];
  p[i]      = (bf16)((t1 * c - t2 * sn) * scale);
  p[i + 32] = (bf16)((t2 * c + t1 * sn) * scale);
}

// ---------------------------------------------------------------------------
// Flash attention, causal, GQA (4 q-heads per kv-head).
// Grid: (Sc/128, Bc*Hc). Block: 256 threads = 8 waves; each wave owns 16 q
// rows. Per 32-wide kv chunk: QK^T via 4 WMMAs, online softmax with VALU
// (dpp8/permlane16) row reductions, P re-layout through LDS (C-frag ->
// A-frag), PV via 4 WMMAs into f32 accumulators. K/V working set per
// (b, kv-head) is 512 KB shared by 64 blocks -> L2-resident (192 MB).
// ---------------------------------------------------------------------------
__global__ __launch_bounds__(256, 2) void attn_kernel(
    const bf16* __restrict__ qb, const bf16* __restrict__ kb,
    const bf16* __restrict__ vT, bf16* __restrict__ attn) {
  __shared__ __align__(16) bf16 pbuf[8][16 * 32];

  const int lane = threadIdx.x & 31;
  const int w    = threadIdx.x >> 5;
  const int ln   = lane & 15;
  const int g    = lane >> 4;
  const int bh   = blockIdx.y;
  const int bIdx = bh >> 5;        // / Hc
  const int h    = bh & (Hc - 1);
  const int hkv  = h >> 2;         // Hc / HKV == 4
  const int qbase = blockIdx.x * 128 + w * 16;

  // Q fragments (rope'd, pre-scaled), A layout: lane = q-row, two 32-K slabs.
  AB aQ0, aQ1;
  {
    const bf16* qp = qb + ((size_t)(bIdx * Sc + qbase + ln)) * Dc + h * HD;
    aQ0.u4[0] = ld4g(qp + g * 16);
    aQ0.u4[1] = ld4g(qp + g * 16 + 8);
    aQ1.u4[0] = ld4g(qp + 32 + g * 16);
    aQ1.u4[1] = ld4g(qp + 32 + g * 16 + 8);
  }

  f32x8 Oacc[4];
#pragma unroll
  for (int t = 0; t < 4; ++t) Oacc[t] = zero8();
  float rM[8], rL[8];
#pragma unroll
  for (int r = 0; r < 8; ++r) { rM[r] = -1e30f; rL[r] = 0.0f; }

  const int kv_end = qbase + 16;           // causal upper bound (exclusive)
  for (int kv0 = 0; kv0 < kv_end; kv0 += 32) {
    // ---- scores: S(16x32) = Q(16x64) @ K^T, two 16x16 tiles --------------
    f32x8 sc[2];
#pragma unroll
    for (int half = 0; half < 2; ++half) {
      const bf16* kp =
          kb + ((size_t)(bIdx * Sc + kv0 + half * 16 + ln)) * DKV + hkv * HD;
      AB b0, b1;  // B-frags: lane = kv column, K-dim = head dim (contiguous)
      b0.u4[0] = ld4g(kp + g * 16);
      b0.u4[1] = ld4g(kp + g * 16 + 8);
      b1.u4[0] = ld4g(kp + 32 + g * 16);
      b1.u4[1] = ld4g(kp + 32 + g * 16 + 8);
      __builtin_prefetch(kp + (size_t)32 * DKV, 0, 1);  // next kv chunk
      f32x8 c = zero8();
      c = wmma_bf16(aQ0.v, b0.v, c);       // d = 0..31
      c = wmma_bf16(aQ1.v, b1.v, c);       // d = 32..63
      sc[half] = c;
    }

    // ---- online softmax (per-row over the 16 lanes sharing the row) ------
#pragma unroll
    for (int r = 0; r < 8; ++r) {
      const int qrow = qbase + r + 8 * g;
      float s0 = sc[0][r];                 // q pre-scaled by 1/sqrt(64)
      float s1 = sc[1][r];
      if (kv0 + ln > qrow)      s0 = -1e30f;
      if (kv0 + 16 + ln > qrow) s1 = -1e30f;
      const float mx    = red_max16(fmaxf(s0, s1));
      const float mnew  = fmaxf(rM[r], mx);
      const float alpha = __expf(rM[r] - mnew);
      const float p0 = __expf(s0 - mnew);
      const float p1 = __expf(s1 - mnew);
      const float rs = red_sum16(p0 + p1);
      rL[r] = rL[r] * alpha + rs;
      rM[r] = mnew;
#pragma unroll
      for (int t = 0; t < 4; ++t) Oacc[t][r] *= alpha;
      // stash P tile in LDS (C layout) for A-frag re-layout
      pbuf[w][(r + 8 * g) * 32 + ln]      = (bf16)p0;
      pbuf[w][(r + 8 * g) * 32 + 16 + ln] = (bf16)p1;
    }

    // ---- P A-fragment from LDS (ds_load_b128 x2, wave-private region) ----
    AB aP;
    aP.u4[0] = *reinterpret_cast<const uint4*>(&pbuf[w][ln * 32 + g * 16]);
    aP.u4[1] = *reinterpret_cast<const uint4*>(&pbuf[w][ln * 32 + g * 16 + 8]);

    // ---- O += P(16x32) @ V(32x64); V stored transposed -> contiguous -----
#pragma unroll
    for (int t = 0; t < 4; ++t) {
      const bf16* vp = vT +
          ((size_t)((bIdx * HKV + hkv) * HD + t * 16 + ln)) * Sc + kv0 + g * 16;
      AB bv;
      bv.u4[0] = ld4g(vp);
      bv.u4[1] = ld4g(vp + 8);
      __builtin_prefetch(vp + 32, 0, 1);   // next kv chunk
      Oacc[t] = wmma_bf16(aP.v, bv.v, Oacc[t]);
    }
  }

  // ---- finalize: O / rowsum, store bf16 (b, s, h*64+dd) --------------------
#pragma unroll
  for (int t = 0; t < 4; ++t)
#pragma unroll
    for (int r = 0; r < 8; ++r) {
      const int srow = qbase + r + 8 * g;
      attn[((size_t)(bIdx * Sc + srow)) * Dc + h * HD + t * 16 + ln] =
          (bf16)(Oacc[t][r] / rL[r]);
    }
}

// ---------------------------------------------------------------------------
// Host-side orchestration.
// ---------------------------------------------------------------------------
extern "C" void kernel_launch(void* const* d_in, const int* in_sizes, int n_in,
                              void* d_out, int out_size, void* d_ws,
                              size_t ws_size, hipStream_t stream) {
  (void)in_sizes; (void)n_in; (void)out_size; (void)ws_size;
  const float* x  = (const float*)d_in[0];
  const float* wq = (const float*)d_in[1];
  const float* wk = (const float*)d_in[2];
  const float* wv = (const float*)d_in[3];
  const float* wo = (const float*)d_in[4];

  char* ws = (char*)d_ws;
  const size_t MB = 1024ull * 1024ull;
  // 60 MB workspace layout (xb region is reused for attn output: all GEMMs
  // reading xb are serialized on `stream` before attn_kernel writes it).
  bf16* xb  = (bf16*)(ws + 0);        // 16 MB  x in bf16  (later: attn out)
  bf16* wqT = (bf16*)(ws + 16 * MB);  //  8 MB
  bf16* wkT = (bf16*)(ws + 24 * MB);  //  2 MB
  bf16* wvT = (bf16*)(ws + 26 * MB);  //  2 MB
  bf16* woT = (bf16*)(ws + 28 * MB);  //  8 MB
  bf16* q_  = (bf16*)(ws + 36 * MB);  // 16 MB
  bf16* k_  = (bf16*)(ws + 52 * MB);  //  4 MB
  bf16* vT  = (bf16*)(ws + 56 * MB);  //  4 MB
  bf16* at  = xb;                     // reuse

  const size_t nx = (size_t)Bc * Sc * Dc;
  cvt_kernel<<<2048, 256, 0, stream>>>(x, xb, nx);
  transpose_cvt_kernel<<<(2048 * 2048) / 256, 256, 0, stream>>>(wq, wqT, 2048, 2048);
  transpose_cvt_kernel<<<(2048 * 512) / 256, 256, 0, stream>>>(wk, wkT, 2048, 512);
  transpose_cvt_kernel<<<(2048 * 512) / 256, 256, 0, stream>>>(wv, wvT, 2048, 512);
  transpose_cvt_kernel<<<(2048 * 2048) / 256, 256, 0, stream>>>(wo, woT, 2048, 2048);

  // QKV projections (M = B*S = 4096)
  gemm_kernel<0><<<dim3(32, 32), 256, 0, stream>>>(xb, wqT, q_, 4096, 2048, 2048);
  gemm_kernel<0><<<dim3(32, 8),  256, 0, stream>>>(xb, wkT, k_, 4096, 512, 2048);
  gemm_kernel<2><<<dim3(32, 8),  256, 0, stream>>>(xb, wvT, vT, 4096, 512, 2048);

  // RoPE (in place); q gets the 1/sqrt(64) attention scale folded in
  rope_kernel<<<(Bc * Sc * Hc * 32) / 256, 256, 0, stream>>>(q_, Hc, 0.125f);
  rope_kernel<<<(Bc * Sc * HKV * 32) / 256, 256, 0, stream>>>(k_, HKV, 1.0f);

  // Flash attention
  attn_kernel<<<dim3(Sc / 128, Bc * Hc), 256, 0, stream>>>(q_, k_, vT, at);

  // Output projection -> f32 d_out
  gemm_kernel<1><<<dim3(32, 32), 256, 0, stream>>>(at, woT, d_out, 4096, 2048, 2048);
}
